// BatchDNC_223338299563
// MI455X (gfx1250) — compile-verified, matrix operand order checked by reading, not run on hardware
//
#include <hip/hip_runtime.h>
#include <math.h>

// ---------------- CDNA5 WMMA types ----------------
typedef __attribute__((ext_vector_type(16))) __bf16    v16bf;
typedef __attribute__((ext_vector_type(8)))  float     v8f;
typedef __attribute__((ext_vector_type(8)))  unsigned  v8u;

#define cBS 16
#define cX  47764
#define cH  128
#define cL  16
#define cVT 3620
#define cW  32
#define cR  8
#define cN  512
#define cKX 48020   // X + W*R
#define cET 395
#define cG  512     // 4*H
#define cLH 2048    // L*H

// RNE f32->bf16 (scalar tail path only)
__device__ __forceinline__ __bf16 f2bf(float f) {
  union { float f; unsigned u; } v; v.f = f;
  unsigned r = (v.u + 0x7FFFu + ((v.u >> 16) & 1u)) >> 16;
  unsigned short s = (unsigned short)r;
  return __builtin_bit_cast(__bf16, s);
}
// two f32 -> dword of two bf16 (truncate) via one v_perm_b32
__device__ __forceinline__ unsigned pk2bf(float lo, float hi) {
  return __builtin_amdgcn_perm(__builtin_bit_cast(unsigned, hi),
                               __builtin_bit_cast(unsigned, lo),
                               0x07060302u);
}
// 16 f32 (two K-groups of 8) -> one v16bf WMMA fragment
__device__ __forceinline__ v16bf mk_frag(float4 a0, float4 a1, float4 a2, float4 a3) {
  v8u u;
  u[0] = pk2bf(a0.x, a0.y); u[1] = pk2bf(a0.z, a0.w);
  u[2] = pk2bf(a1.x, a1.y); u[3] = pk2bf(a1.z, a1.w);
  u[4] = pk2bf(a2.x, a2.y); u[5] = pk2bf(a2.z, a2.w);
  u[6] = pk2bf(a3.x, a3.y); u[7] = pk2bf(a3.z, a3.w);
  return __builtin_bit_cast(v16bf, u);
}
__device__ __forceinline__ float sigf(float x) { return 1.0f / (1.0f + __expf(-x)); }
__device__ __forceinline__ float logsigf(float x) {
  return (x >= 0.0f) ? -log1pf(__expf(-x)) : (x - log1pf(__expf(x)));
}

// =====================================================================
// Generic M=16 GEMM: C(16xN) += A(16xK,f32) * B, bf16 WMMA, K-split with
// f32 atomic accumulation.  B_NK: B[n*ldb+k] (X @ W^T).  else B[k*ldb+n].
// grid = (ceil(N/16), kChunks), block = 32 threads (1 wave).
// =====================================================================
template<bool B_NK>
__global__ void wmma_gemm_k(const float* __restrict__ A, int lda,
                            const float* __restrict__ B, int ldb,
                            float* __restrict__ C, int ldc,
                            int N, int K, int kChunk)
{
  const int lane = threadIdx.x;          // 0..31
  const int m    = lane & 15;            // A row for this lane
  const int hi   = lane >> 4;            // K-half selector
  const int n    = blockIdx.x * 16 + (lane & 15);
  const int k0   = blockIdx.y * kChunk;
  int kend = k0 + kChunk; if (kend > K) kend = K;
  int len  = kend - k0;   if (len < 0)  len = 0;
  const int kfull = k0 + (len & ~31);    // full 32-wide chunks

  v8f acc = {};
  const float* Arow = A + (long)m * lda;

  // ---- hot loop: unguarded b128 loads + v_perm packing ----
  for (int kb = k0; kb < kfull; kb += 32) {
    const int kg = kb + hi * 8;
    __builtin_prefetch(&Arow[kg + 256], 0, 1);
    const float4 a0 = *(const float4*)&Arow[kg];
    const float4 a1 = *(const float4*)&Arow[kg + 4];
    const float4 a2 = *(const float4*)&Arow[kg + 16];
    const float4 a3 = *(const float4*)&Arow[kg + 20];
    const v16bf av = mk_frag(a0, a1, a2, a3);

    float4 b0 = {}, b1 = {}, b2 = {}, b3 = {};
    if (n < N) {
      if (B_NK) {
        const float* Brow = B + (long)n * ldb;
        b0 = *(const float4*)&Brow[kg];
        b1 = *(const float4*)&Brow[kg + 4];
        b2 = *(const float4*)&Brow[kg + 16];
        b3 = *(const float4*)&Brow[kg + 20];
      } else {
        const float* Bc = B + n;
        b0.x = Bc[(long)(kg + 0) * ldb];  b0.y = Bc[(long)(kg + 1) * ldb];
        b0.z = Bc[(long)(kg + 2) * ldb];  b0.w = Bc[(long)(kg + 3) * ldb];
        b1.x = Bc[(long)(kg + 4) * ldb];  b1.y = Bc[(long)(kg + 5) * ldb];
        b1.z = Bc[(long)(kg + 6) * ldb];  b1.w = Bc[(long)(kg + 7) * ldb];
        b2.x = Bc[(long)(kg + 16) * ldb]; b2.y = Bc[(long)(kg + 17) * ldb];
        b2.z = Bc[(long)(kg + 18) * ldb]; b2.w = Bc[(long)(kg + 19) * ldb];
        b3.x = Bc[(long)(kg + 20) * ldb]; b3.y = Bc[(long)(kg + 21) * ldb];
        b3.z = Bc[(long)(kg + 22) * ldb]; b3.w = Bc[(long)(kg + 23) * ldb];
      }
    }
    const v16bf bv = mk_frag(b0, b1, b2, b3);
    acc = __builtin_amdgcn_wmma_f32_16x16x32_bf16(false, av, false, bv,
                                                  (short)0, acc, false, false);
  }

  // ---- guarded scalar tail (K remainder < 32) ----
  for (int kb = kfull; kb < kend; kb += 32) {
    v16bf av, bv;
#pragma unroll
    for (int j = 0; j < 8; ++j) {
      const int ka = kb + hi * 8 + j;
      const int kc = ka + 16;
      av[j]     = f2bf((ka < K) ? Arow[ka] : 0.0f);
      av[j + 8] = f2bf((kc < K) ? Arow[kc] : 0.0f);
      float bb0 = 0.0f, bb1 = 0.0f;
      if (n < N) {
        if (ka < K) bb0 = B_NK ? B[(long)n * ldb + ka] : B[(long)ka * ldb + n];
        if (kc < K) bb1 = B_NK ? B[(long)n * ldb + kc] : B[(long)kc * ldb + n];
      }
      bv[j]     = f2bf(bb0);
      bv[j + 8] = f2bf(bb1);
    }
    acc = __builtin_amdgcn_wmma_f32_16x16x32_bf16(false, av, false, bv,
                                                  (short)0, acc, false, false);
  }

  if (n < N) {
#pragma unroll
    for (int v = 0; v < 8; ++v) {
      const int mo = v + 8 * hi;             // C layout: M = vgpr + 8*(lane>=16)
      atomicAdd(&C[(long)mo * ldc + n], acc[v]);
    }
  }
}

// Wave-level 16x16 tile accumulate over K (multiple of 32), B stored NxK.
// A may live in LDS (generic pointer -> ds_load_b128).
__device__ __forceinline__ v8f wmma_nk_acc(const float* A, int lda,
                                           const float* B, int ldb,
                                           int nBase, int K, int lane, v8f acc)
{
  const int m  = lane & 15;
  const int hi = lane >> 4;
  const int n  = nBase + (lane & 15);
  const float* Arow = A + m * lda;
  const float* Brow = B + (long)n * ldb;
  for (int kb = 0; kb < K; kb += 32) {
    const int kg = kb + hi * 8;
    const v16bf av = mk_frag(*(const float4*)&Arow[kg],
                             *(const float4*)&Arow[kg + 4],
                             *(const float4*)&Arow[kg + 16],
                             *(const float4*)&Arow[kg + 20]);
    const v16bf bv = mk_frag(*(const float4*)&Brow[kg],
                             *(const float4*)&Brow[kg + 4],
                             *(const float4*)&Brow[kg + 16],
                             *(const float4*)&Brow[kg + 20]);
    acc = __builtin_amdgcn_wmma_f32_16x16x32_bf16(false, av, false, bv,
                                                  (short)0, acc, false, false);
  }
  return acc;
}

// =====================================================================
// Stage A: batch-norm over batch dim + concat read vectors -> x_t
// =====================================================================
__global__ void prep_xt_kernel(const float* __restrict__ x,
                               const float* __restrict__ lrv,
                               float* __restrict__ xt)
{
  const int j = blockIdx.x * blockDim.x + threadIdx.x;
  if (j >= cKX) return;
  if (j < cX) {
    float mu = 0.0f;
#pragma unroll
    for (int b = 0; b < cBS; ++b) mu += x[(long)b * cX + j];
    mu *= (1.0f / cBS);
    float var = 0.0f;
#pragma unroll
    for (int b = 0; b < cBS; ++b) {
      float d = x[(long)b * cX + j] - mu;
      var += d * d;
    }
    var *= (1.0f / cBS);
    const float inv = 1.0f / sqrtf(var + 0.001f);
#pragma unroll
    for (int b = 0; b < cBS; ++b)
      xt[(long)b * cKX + j] = (x[(long)b * cX + j] - mu) * inv;
  } else {
    const int t = j - cX;  // last_read_vec (BS,W,R) flattened row-major
#pragma unroll
    for (int b = 0; b < cBS; ++b)
      xt[(long)b * cKX + j] = lrv[b * (cW * cR) + t];
  }
}

// =====================================================================
// Stage B: 16-layer LSTM step.  One block of 1024 threads = 32 waves;
// wave w owns gate-column tile [16w, 16w+16).  Hidden state in LDS.
// G0 already holds x_t @ W_ih0^T (atomic-accumulated, no biases).
// =====================================================================
__global__ void __launch_bounds__(1024)
lstm_kernel(const float* __restrict__ G0,
            const float* __restrict__ h0, const float* __restrict__ c0,
            const float* __restrict__ b_ih0,
            const float* __restrict__ W_ihr, const float* __restrict__ b_ihr,
            const float* __restrict__ W_hh,  const float* __restrict__ b_hh,
            float* __restrict__ fh)
{
  __shared__ float s_gates[cBS * cG];   // 16x512
  __shared__ float s_h[cBS * cH];       // 16x128
  const int tid  = threadIdx.x;
  const int wave = tid >> 5;
  const int lane = tid & 31;
  const int hi   = lane >> 4;
  const int n    = wave * 16 + (lane & 15);

  // ---- layer 0: + h0[0] @ W_hh[0]^T, + biases ----
  {
    v8f acc = {};
    acc = wmma_nk_acc(h0, cH, W_hh, cH, wave * 16, cH, lane, acc);
#pragma unroll
    for (int v = 0; v < 8; ++v) {
      const int mo = v + 8 * hi;
      s_gates[mo * cG + n] = acc[v] + G0[mo * cG + n] + b_ih0[n] + b_hh[n];
    }
  }
  __syncthreads();
  // nonlinearity, layer 0
  for (int e = tid; e < cBS * cH; e += 1024) {
    const int b = e >> 7, j = e & 127;
    const float gi = s_gates[b * cG + j];
    const float gf = s_gates[b * cG + cH + j];
    const float gg = s_gates[b * cG + 2 * cH + j];
    const float go = s_gates[b * cG + 3 * cH + j];
    const float c  = sigf(gf) * c0[b * cH + j] + sigf(gi) * tanhf(gg);
    const float h  = sigf(go) * tanhf(c);
    s_h[e] = h;
    fh[b * cLH + j] = h;   // l = 0
  }
  __syncthreads();

  // ---- layers 1..15 ----
  for (int l = 1; l < cL; ++l) {
    v8f acc = {};
    acc = wmma_nk_acc((const float*)s_h, cH, W_ihr + (long)(l - 1) * cG * cH,
                      cH, wave * 16, cH, lane, acc);
    acc = wmma_nk_acc(h0 + (long)l * cBS * cH, cH, W_hh + (long)l * cG * cH,
                      cH, wave * 16, cH, lane, acc);
#pragma unroll
    for (int v = 0; v < 8; ++v) {
      const int mo = v + 8 * hi;
      s_gates[mo * cG + n] = acc[v] + b_ihr[(l - 1) * cG + n] + b_hh[l * cG + n];
    }
    __syncthreads();
    for (int e = tid; e < cBS * cH; e += 1024) {
      const int b = e >> 7, j = e & 127;
      const float gi = s_gates[b * cG + j];
      const float gf = s_gates[b * cG + cH + j];
      const float gg = s_gates[b * cG + 2 * cH + j];
      const float go = s_gates[b * cG + 3 * cH + j];
      const float c  = sigf(gf) * c0[(long)l * cBS * cH + b * cH + j]
                     + sigf(gi) * tanhf(gg);
      const float h  = sigf(go) * tanhf(c);
      s_h[e] = h;
      fh[b * cLH + l * cH + j] = h;
    }
    __syncthreads();
  }
}

// =====================================================================
// Stage D: DNC memory machinery. One block (512 threads) per batch.
// =====================================================================
__device__ __forceinline__ float blk_max(float v, float* red, int tid) {
  red[tid] = v; __syncthreads();
  for (int o = 256; o > 0; o >>= 1) {
    if (tid < o) red[tid] = fmaxf(red[tid], red[tid + o]);
    __syncthreads();
  }
  float r = red[0]; __syncthreads(); return r;
}
__device__ __forceinline__ float blk_sum(float v, float* red, int tid) {
  red[tid] = v; __syncthreads();
  for (int o = 256; o > 0; o >>= 1) {
    if (tid < o) red[tid] += red[tid + o];
    __syncthreads();
  }
  float r = red[0]; __syncthreads(); return r;
}

__global__ void __launch_bounds__(512)
dnc_kernel(const float* __restrict__ memory, const float* __restrict__ lrw,
           const float* __restrict__ lww,    const float* __restrict__ prec,
           const float* __restrict__ linkage,const float* __restrict__ lusage,
           const float* __restrict__ iibuf,  float* __restrict__ rvbuf)
{
  const int b   = blockIdx.x;
  const int tid = threadIdx.x;
  const int n   = tid;

  __shared__ float s_rk[cW * cR];    // read keys [w*R+r]
  __shared__ float s_wk[cW], s_er[cW], s_wv[cW];
  __shared__ float s_rs[cR], s_fg[cR], s_rkn[cR];
  __shared__ float s_rm[cR][3];
  __shared__ float s_scal[4];        // 0:wstr 1:agate 2:wgate 3:wk_norm
  __shared__ float s_key[cN];
  __shared__ int   s_idx[cN];
  __shared__ float s_ww[cN], s_alloc[cN], s_red[cN];
  __shared__ float s_lrw[cN * cR];   // 16 KB
  __shared__ float s_simr[cN * cR];  // 16 KB
  __shared__ float s_rv[cW * cR];

  const float* ii = iibuf + (long)b * cET;

  // ---- parse interface vector ----
  if (tid < 256)            s_rk[tid] = ii[tid];
  else if (tid < 264)       s_rs[tid - 256] = 1.0f - logsigf(ii[tid]);
  else if (tid < 296)       s_wk[tid - 264] = ii[tid];
  else if (tid == 296)      s_scal[0] = 1.0f - logsigf(ii[296]);
  else if (tid < 329)       s_er[tid - 297] = sigf(ii[tid]);
  else if (tid < 361)       s_wv[tid - 329] = ii[tid];
  else if (tid < 369)       s_fg[tid - 361] = sigf(ii[tid]);
  else if (tid == 369)      s_scal[1] = sigf(ii[369]);
  else if (tid == 370)      s_scal[2] = sigf(ii[370]);
  else if (tid < 379) {
    const int r = tid - 371;
    float a0 = ii[371 + r * 3], a1 = ii[372 + r * 3], a2 = ii[373 + r * 3];
    float mx = fmaxf(a0, fmaxf(a1, a2));
    float e0 = __expf(a0 - mx), e1 = __expf(a1 - mx), e2 = __expf(a2 - mx);
    float s = e0 + e1 + e2;
    s_rm[r][0] = e0 / s; s_rm[r][1] = e1 / s; s_rm[r][2] = e2 / s;
  }
  for (int e = tid; e < cN * cR; e += 512) s_lrw[e] = lrw[(long)b * cN * cR + e];
  __syncthreads();
  if (tid < cR) {            // read-key column norms
    float s = 0.0f;
    for (int w = 0; w < cW; ++w) { float v = s_rk[w * cR + tid]; s += v * v; }
    s_rkn[tid] = sqrtf(s);
  } else if (tid == cR) {    // write-key norm
    float s = 0.0f;
    for (int w = 0; w < cW; ++w) { float v = s_wk[w]; s += v * v; }
    s_scal[3] = sqrtf(s);
  }
  __syncthreads();

  // ---- retention / usage ----
  float ret = 1.0f;
#pragma unroll
  for (int r = 0; r < cR; ++r) ret *= 1.0f - s_fg[r] * s_lrw[n * cR + r];
  {
    const float lu = lusage[b * cN + n], lw = lww[b * cN + n];
    s_key[n] = (lu + lw - lu * lw) * ret;
    s_idx[n] = n;
  }

  // ---- bitonic argsort (ascending by usage) ----
  for (int k2 = 2; k2 <= cN; k2 <<= 1) {
    for (int j = k2 >> 1; j > 0; j >>= 1) {
      __syncthreads();
      const int ixj = tid ^ j;
      if (ixj > tid) {
        const bool up = ((tid & k2) == 0);
        float ka = s_key[tid], kb2 = s_key[ixj];
        if ((ka > kb2) == up) {
          s_key[tid] = kb2; s_key[ixj] = ka;
          int t = s_idx[tid]; s_idx[tid] = s_idx[ixj]; s_idx[ixj] = t;
        }
      }
    }
  }
  __syncthreads();

  // ---- exclusive prefix product over sorted usage ----
  s_red[tid] = s_key[tid];
  __syncthreads();
  for (int off = 1; off < cN; off <<= 1) {
    float cur = s_red[tid];
    float pre = (tid >= off) ? s_red[tid - off] : 1.0f;
    __syncthreads();
    s_red[tid] = cur * pre;
    __syncthreads();
  }
  {
    const float cp = (tid == 0) ? 1.0f : s_red[tid - 1];
    s_alloc[s_idx[tid]] = (1.0f - s_key[tid]) * cp;
  }
  __syncthreads();
  s_key[tid] = prec[b * cN + tid];   // reuse s_key for precedence
  __syncthreads();

  // ---- content write addressing ----
  const float* mrow = memory + ((long)b * cN + n) * cW;
  float dot = 0.0f, nrm = 0.0f;
#pragma unroll
  for (int w = 0; w < cW; ++w) {
    const float mv = mrow[w];
    dot += mv * s_wk[w]; nrm += mv * mv;
  }
  const float simw = dot / fmaxf(sqrtf(nrm) * s_scal[3], 1e-8f);
  const float zw = s_scal[0] * simw;
  {
    const float mx = blk_max(zw, s_red, tid);
    const float ex = __expf(zw - mx);
    const float sm = blk_sum(ex, s_red, tid);
    const float cwn = ex / sm;
    const float ag = s_scal[1], wg = s_scal[2];
    s_ww[n] = wg * (ag * s_alloc[n] + (1.0f - ag) * cwn);
  }
  __syncthreads();
  const float ww_n = s_ww[n];

  // ---- memory write (row kept in registers) ----
  float m2row[cW];
#pragma unroll
  for (int w = 0; w < cW; ++w)
    m2row[w] = mrow[w] * (1.0f - ww_n * s_er[w]) + ww_n * s_wv[w];

  // ---- forward / backward weights via on-the-fly linkage ----
  float fwv[cR], bwv[cR];
#pragma unroll
  for (int r = 0; r < cR; ++r) { fwv[r] = 0.0f; bwv[r] = 0.0f; }
  {
    const float prec_n = s_key[n];
    const float* lkrow = linkage + ((long)b * cN + n) * cN;
    for (int j = 0; j < cN; ++j) {
      const float ww_j = s_ww[j];
      const float pj   = s_key[j];
      const float lnj  = lkrow[j];
      const float ljn  = linkage[((long)b * cN + j) * cN + n];
      float lk_nj = (n == j) ? 0.0f : (1.0f - ww_n - ww_j) * lnj + ww_n * pj;
      float lk_jn = (n == j) ? 0.0f : (1.0f - ww_j - ww_n) * ljn + ww_j * prec_n;
#pragma unroll
      for (int r = 0; r < cR; ++r) {
        const float lw = s_lrw[j * cR + r];
        fwv[r] += lk_nj * lw;
        bwv[r] += lk_jn * lw;
      }
    }
  }

  // ---- content read addressing ----
  float n2 = 0.0f;
#pragma unroll
  for (int w = 0; w < cW; ++w) n2 += m2row[w] * m2row[w];
  const float m2n = sqrtf(n2);
#pragma unroll
  for (int r = 0; r < cR; ++r) {
    float d = 0.0f;
#pragma unroll
    for (int w = 0; w < cW; ++w) d += m2row[w] * s_rk[w * cR + r];
    const float sim = d / fmaxf(m2n * s_rkn[r], 1e-8f);
    s_simr[n * cR + r] = s_rs[r] * sim;
  }
  __syncthreads();
  for (int r = 0; r < cR; ++r) {   // softmax over n, each r
    const float z  = s_simr[n * cR + r];
    const float mx = blk_max(z, s_red, tid);
    const float ex = __expf(z - mx);
    const float sm = blk_sum(ex, s_red, tid);
    s_simr[n * cR + r] = ex / sm;
    __syncthreads();
  }

  // ---- read weights + read vectors ----
  if (tid < cW * cR) s_rv[tid] = 0.0f;
  __syncthreads();
  float rwv[cR];
#pragma unroll
  for (int r = 0; r < cR; ++r)
    rwv[r] = s_rm[r][0] * bwv[r] + s_rm[r][1] * s_simr[n * cR + r]
           + s_rm[r][2] * fwv[r];
#pragma unroll
  for (int w = 0; w < cW; ++w) {
    const float mv = m2row[w];
#pragma unroll
    for (int r = 0; r < cR; ++r) atomicAdd(&s_rv[w * cR + r], mv * rwv[r]);
  }
  __syncthreads();
  if (tid < cW * cR) rvbuf[b * (cW * cR) + tid] = s_rv[tid];
}

// =====================================================================
// Host side
// =====================================================================
extern "C" void kernel_launch(void* const* d_in, const int* in_sizes, int n_in,
                              void* d_out, int out_size, void* d_ws, size_t ws_size,
                              hipStream_t stream)
{
  const float* x       = (const float*)d_in[0];
  const float* memory  = (const float*)d_in[1];
  const float* lrw     = (const float*)d_in[2];
  const float* lww     = (const float*)d_in[3];
  const float* prec    = (const float*)d_in[4];
  const float* linkage = (const float*)d_in[5];
  const float* lusage  = (const float*)d_in[6];
  const float* lrv     = (const float*)d_in[7];
  const float* h0      = (const float*)d_in[8];
  const float* c0      = (const float*)d_in[9];
  const float* W_ih0   = (const float*)d_in[10];
  const float* b_ih0   = (const float*)d_in[11];
  const float* W_ihr   = (const float*)d_in[12];
  const float* b_ihr   = (const float*)d_in[13];
  const float* W_hh    = (const float*)d_in[14];
  const float* b_hh    = (const float*)d_in[15];
  const float* W_y     = (const float*)d_in[16];
  const float* W_E     = (const float*)d_in[17];
  const float* W_r     = (const float*)d_in[18];
  float* out = (float*)d_out;
  float* ws  = (float*)d_ws;

  const long OFF_XT = 0;                              // 16*48020
  const long OFF_G0 = OFF_XT + (long)cBS * cKX;       // 16*512
  const long OFF_FH = OFF_G0 + (long)cBS * cG;        // 16*2048
  const long OFF_II = OFF_FH + (long)cBS * cLH;       // 16*395
  const long OFF_RV = OFF_II + (long)cBS * cET;       // 16*256
  float* xt  = ws + OFF_XT;
  float* G0  = ws + OFF_G0;
  float* fh  = ws + OFF_FH;
  float* iib = ws + OFF_II;
  float* rvb = ws + OFF_RV;

  // zero the atomic-accumulated buffers
  hipMemsetAsync(out, 0, (size_t)cBS * cVT * sizeof(float), stream);
  hipMemsetAsync(G0,  0, (size_t)cBS * cG  * sizeof(float), stream);
  hipMemsetAsync(iib, 0, (size_t)cBS * cET * sizeof(float), stream);

  // Stage A: batch-norm + concat
  prep_xt_kernel<<<(cKX + 255) / 256, 256, 0, stream>>>(x, lrv, xt);

  // Stage B1: G0 = x_t @ W_ih0^T  (K = 48020 split into 64 chunks of 768)
  wmma_gemm_k<true><<<dim3(cG / 16, 64), 32, 0, stream>>>(
      xt, cKX, W_ih0, cKX, G0, cG, cG, cKX, 768);

  // Stage B2: full 16-layer LSTM step
  lstm_kernel<<<1, 1024, 0, stream>>>(G0, h0, c0, b_ih0, W_ihr, b_ihr,
                                      W_hh, b_hh, fh);

  // Stage C: vt -> out,  ii -> iib
  wmma_gemm_k<false><<<dim3((cVT + 15) / 16, 8), 32, 0, stream>>>(
      fh, cLH, W_y, cVT, out, cVT, cVT, cLH, 256);
  wmma_gemm_k<false><<<dim3((cET + 15) / 16, 2), 32, 0, stream>>>(
      fh, cLH, W_E, cET, iib, cET, cET, cLH, 1024);

  // Stage D: DNC memory machinery (one block per batch)
  dnc_kernel<<<cBS, 512, 0, stream>>>(memory, lrw, lww, prec, linkage,
                                      lusage, iib, rvb);

  // Stage E: out += rv @ W_r
  wmma_gemm_k<false><<<dim3((cVT + 15) / 16, 1), 32, 0, stream>>>(
      rvb, cW * cR, W_r, cVT, out, cVT, cVT, cW * cR, cW * cR);
}
